// SyntaxTransformerEncoder_76330158784745
// MI455X (gfx1250) — matrix operand order repaired and checked by reading, not compiled, not run on hardware
//
#include <hip/hip_runtime.h>

// ---------------------------------------------------------------------------
// Types for CDNA5 WMMA (gfx1250, wave32)
// ---------------------------------------------------------------------------
typedef __bf16 bfloat;
typedef __bf16  v16bf __attribute__((ext_vector_type(16)));
typedef __bf16  v8bf  __attribute__((ext_vector_type(8)));
typedef float   v8f   __attribute__((ext_vector_type(8)));

#define FLAG_TRANSB 1
#define FLAG_RELU   2
#define FLAG_OUTBF  4

// CDNA5 async global->LDS copy (ASYNCcnt-tracked, no VGPR staging).
__device__ __forceinline__ void async_copy_b128(unsigned lds_off, const void* gaddr)
{
    asm volatile("global_load_async_to_lds_b128 %0, %1, off"
                 :: "v"(lds_off), "v"(gaddr) : "memory");
}
__device__ __forceinline__ void wait_async_le3() { asm volatile("s_wait_asynccnt 0x3" ::: "memory"); }
__device__ __forceinline__ void wait_async_le2() { asm volatile("s_wait_asynccnt 0x2" ::: "memory"); }
__device__ __forceinline__ void wait_async_le0() { asm volatile("s_wait_asynccnt 0x0" ::: "memory"); }

// ---------------------------------------------------------------------------
// Generic bf16 WMMA GEMM:  C = act(alpha * A*B + bias + residual)
//   A: [M,K] bf16 row-major (lda), B: [K,N] bf16 (ldb)  or  [N,K] if TRANSB
//   batched via blockIdx.z = z1*nz2+z2 with per-matrix (Z1,Z2) strides.
//   Block tile 128x64, K-step 32, 256 threads = 8 waves.
//   Each wave: 32x32 outputs = 2 A-frags x 2 B-frags -> 4 WMMAs (frag reuse).
//   Double-buffered LDS fed by async global->LDS loads.
// ---------------------------------------------------------------------------
template <bool TRANSB>
__global__ __launch_bounds__(256)
void wmma_gemm_kernel(const bfloat* __restrict__ A, const bfloat* __restrict__ Bm,
                      const float* __restrict__ bias, const float* __restrict__ residual,
                      void* __restrict__ Cout,
                      int K, int lda, int ldb, int ldc, int ldr,
                      long long aZ1, long long aZ2, long long bZ1, long long bZ2,
                      long long cZ1, long long cZ2,
                      int nz2, float alpha, int flags)
{
    __shared__ __align__(16) bfloat As[2][128][40];  // 80B rows -> 16B aligned subrows
    __shared__ __align__(16) bfloat Bs[2][64][40];   // stored [n][k]

    const int z  = blockIdx.z;
    const int z1 = z / nz2, z2 = z - z1 * nz2;
    const bfloat* Ab = A  + z1 * aZ1 + z2 * aZ2;
    const bfloat* Bb = Bm + z1 * bZ1 + z2 * bZ2;
    const long long coff = z1 * cZ1 + z2 * cZ2;

    const int m0   = blockIdx.y * 128;
    const int n0   = blockIdx.x * 64;
    const int tid  = threadIdx.x;
    const int lane = tid & 31;
    const int wave = tid >> 5;
    const int wm   = wave >> 1;      // 0..3  (32-row strip)
    const int wn   = wave & 1;       // 0..1  (32-col strip)

    // loader indices
    const int lrow = tid >> 2;             // 0..63
    const int lcg  = (tid & 3) << 3;       // 0,8,16,24
    const int bkk  = tid >> 3;             // 0..31  (!TRANSB)
    const int bng  = (tid & 7) << 3;       // 0..56  (!TRANSB)

    v8f acc00 = {}, acc01 = {}, acc10 = {}, acc11 = {};

    // ---- tile issue: A (2 halves) via async copy; B async if TRANSB else staged
    auto issue_tile = [&](int k0, int buf) {
        async_copy_b128((unsigned)(size_t)&As[buf][lrow][lcg],
                        Ab + (long long)(m0 + lrow) * lda + k0 + lcg);
        async_copy_b128((unsigned)(size_t)&As[buf][lrow + 64][lcg],
                        Ab + (long long)(m0 + lrow + 64) * lda + k0 + lcg);
        if (TRANSB) {
            async_copy_b128((unsigned)(size_t)&Bs[buf][lrow][lcg],
                            Bb + (long long)(n0 + lrow) * ldb + k0 + lcg);
        } else {
            v8bf val = *(const v8bf*)(Bb + (long long)(k0 + bkk) * ldb + n0 + bng);
#pragma unroll
            for (int i = 0; i < 8; ++i) Bs[buf][bng + i][bkk] = val[i];
        }
    };

    issue_tile(0, 0);
    int buf = 0;
    for (int k0 = 0; k0 < K; k0 += 32, buf ^= 1) {
        const bool hasNext = (k0 + 32) < K;
        if (hasNext) issue_tile(k0 + 32, buf ^ 1);
        // wait for *this* buffer's async ops (in-order completion per thread)
        if (TRANSB) { if (hasNext) wait_async_le3(); else wait_async_le0(); }
        else        { if (hasNext) wait_async_le2(); else wait_async_le0(); }
        __syncthreads();

        // ---- A fragments (ISA 16-bit 16x32 layout) ----
        const int arow0 = wm * 32 + (lane & 15);
        const int kbase = (lane >= 16) ? 8 : 0;
        v16bf af0, af1;
        {
            v8bf lo = *(const v8bf*)&As[buf][arow0][kbase];
            v8bf hi = *(const v8bf*)&As[buf][arow0][kbase + 16];
#pragma unroll
            for (int i = 0; i < 8; ++i) { af0[i] = lo[i]; af0[i + 8] = hi[i]; }
        }
        {
            v8bf lo = *(const v8bf*)&As[buf][arow0 + 16][kbase];
            v8bf hi = *(const v8bf*)&As[buf][arow0 + 16][kbase + 16];
#pragma unroll
            for (int i = 0; i < 8; ++i) { af1[i] = lo[i]; af1[i + 8] = hi[i]; }
        }
        // ---- B fragments: N = lane&15, K = (lane>=16?8:0)+e (+16 for e>=8) ----
        const int kb    = (lane >= 16) ? 8 : 0;
        const int bcol0 = wn * 32 + (lane & 15);
        v16bf bf0, bf1;
        {
            v8bf lo = *(const v8bf*)&Bs[buf][bcol0][kb];
            v8bf hi = *(const v8bf*)&Bs[buf][bcol0][kb + 16];
#pragma unroll
            for (int i = 0; i < 8; ++i) { bf0[i] = lo[i]; bf0[i + 8] = hi[i]; }
        }
        {
            v8bf lo = *(const v8bf*)&Bs[buf][bcol0 + 16][kb];
            v8bf hi = *(const v8bf*)&Bs[buf][bcol0 + 16][kb + 16];
#pragma unroll
            for (int i = 0; i < 8; ++i) { bf1[i] = lo[i]; bf1[i + 8] = hi[i]; }
        }
        acc00 = __builtin_amdgcn_wmma_f32_16x16x32_bf16(false, af0, false, bf0,
                                                        (short)0, acc00, false, false);
        acc01 = __builtin_amdgcn_wmma_f32_16x16x32_bf16(false, af0, false, bf1,
                                                        (short)0, acc01, false, false);
        acc10 = __builtin_amdgcn_wmma_f32_16x16x32_bf16(false, af1, false, bf0,
                                                        (short)0, acc10, false, false);
        acc11 = __builtin_amdgcn_wmma_f32_16x16x32_bf16(false, af1, false, bf1,
                                                        (short)0, acc11, false, false);
        __syncthreads();   // protect buffer reuse two iterations ahead
    }

    // ---- epilogue (C layout: N=lane&15, M=vgpr + 8*(lane>=16)) ----
    const int rbase = m0 + wm * 32 + ((lane >= 16) ? 8 : 0);
    const int cbase = n0 + wn * 32 + (lane & 15);
#pragma unroll
    for (int ti = 0; ti < 2; ++ti) {
#pragma unroll
        for (int tj = 0; tj < 2; ++tj) {
            const v8f& acc = ti ? (tj ? acc11 : acc10) : (tj ? acc01 : acc00);
#pragma unroll
            for (int v = 0; v < 8; ++v) {
                const int grow = rbase + ti * 16 + v;
                const int gcol = cbase + tj * 16;
                float val = acc[v] * alpha;
                if (bias)     val += bias[gcol];
                if (residual) val += residual[(long long)grow * ldr + gcol];
                if (flags & FLAG_RELU) val = fmaxf(val, 0.f);
                const long long cidx = coff + (long long)grow * ldc + gcol;
                if (flags & FLAG_OUTBF) ((bfloat*)Cout)[cidx] = (bfloat)val;
                else                    ((float*)Cout)[cidx]  = val;
            }
        }
    }
}

// ---------------------------------------------------------------------------
// Block-wide 2-value reduction helper (256 threads, wave32)
// ---------------------------------------------------------------------------
__device__ __forceinline__ void blockReduce2(float& s, float& q, float* red)
{
    __syncthreads();
#pragma unroll
    for (int o = 16; o; o >>= 1) { s += __shfl_xor(s, o, 32); q += __shfl_xor(q, o, 32); }
    const int wave = threadIdx.x >> 5, lane = threadIdx.x & 31;
    if (lane == 0) { red[wave] = s; red[wave + 8] = q; }
    __syncthreads();
    float ss = 0.f, qq = 0.f;
#pragma unroll
    for (int i = 0; i < 8; ++i) { ss += red[i]; qq += red[i + 8]; }
    s = ss; q = qq;
}

// ---------------------------------------------------------------------------
// LayerNorm (optionally two chained LNs); D fixed at 512; out fp32 or bf16
// ---------------------------------------------------------------------------
__global__ __launch_bounds__(256)
void ln_kernel(const float* __restrict__ x,
               const float* __restrict__ g1, const float* __restrict__ b1,
               const float* __restrict__ g2, const float* __restrict__ b2,
               void* __restrict__ out, int outBf16)
{
    __shared__ float red[16];
    const int r = blockIdx.x, tid = threadIdx.x;
    const float* xr = x + (long long)r * 512;
    float a = xr[tid], c = xr[tid + 256];
    float s = a + c, q = a * a + c * c;
    blockReduce2(s, q, red);
    float m = s * (1.f / 512.f);
    float rs = rsqrtf(q * (1.f / 512.f) - m * m + 1e-5f);
    float y0 = (a - m) * rs * g1[tid]       + b1[tid];
    float y1 = (c - m) * rs * g1[tid + 256] + b1[tid + 256];
    if (g2) {
        s = y0 + y1; q = y0 * y0 + y1 * y1;
        blockReduce2(s, q, red);
        m  = s * (1.f / 512.f);
        rs = rsqrtf(q * (1.f / 512.f) - m * m + 1e-5f);
        y0 = (y0 - m) * rs * g2[tid]       + b2[tid];
        y1 = (y1 - m) * rs * g2[tid + 256] + b2[tid + 256];
    }
    const long long base = (long long)r * 512;
    if (outBf16) {
        ((bfloat*)out)[base + tid]       = (bfloat)y0;
        ((bfloat*)out)[base + tid + 256] = (bfloat)y1;
    } else {
        ((float*)out)[base + tid]       = y0;
        ((float*)out)[base + tid + 256] = y1;
    }
}

// ---------------------------------------------------------------------------
// Embedding: gathers + syntax MLP(3->128 LN relu ->512) + sinusoidal PE + LN
// ---------------------------------------------------------------------------
__global__ __launch_bounds__(256)
void embed_kernel(const int* __restrict__ tok, const int* __restrict__ ptag,
                  const int* __restrict__ etype, const float* __restrict__ sfeat,
                  const float* __restrict__ tok_emb, const float* __restrict__ ptag_emb,
                  const float* __restrict__ elem_emb,
                  const float* __restrict__ w1, const float* __restrict__ b1,
                  const float* __restrict__ lg, const float* __restrict__ lb,
                  const float* __restrict__ w2, const float* __restrict__ b2v,
                  const float* __restrict__ eg, const float* __restrict__ eb,
                  float* __restrict__ xout)
{
    __shared__ float h1[128];
    __shared__ float red[16];
    const int t = blockIdx.x;          // token index in [0, B*S)
    const int spos = t & 1023;
    const int tid = threadIdx.x;
    const int tk = tok[t], pt = ptag[t], et = etype[t];
    const float sf0 = sfeat[t * 3 + 0], sf1 = sfeat[t * 3 + 1], sf2 = sfeat[t * 3 + 2];

    // syntax MLP hidden (128), LN over 128, relu
    float hv = 0.f;
    if (tid < 128)
        hv = sf0 * w1[tid] + sf1 * w1[128 + tid] + sf2 * w1[256 + tid] + b1[tid];
    float s = (tid < 128) ? hv : 0.f, q = (tid < 128) ? hv * hv : 0.f;
    blockReduce2(s, q, red);
    float m = s * (1.f / 128.f);
    float rs = rsqrtf(q * (1.f / 128.f) - m * m + 1e-5f);
    if (tid < 128) h1[tid] = fmaxf((hv - m) * rs * lg[tid] + lb[tid], 0.f);
    __syncthreads();

    const float SQ = 22.62741699796952f;           // sqrt(512)
    const float LG = 9.210340371976184f / 512.f;   // ln(10000)/512
    float xv[2];
#pragma unroll
    for (int e = 0; e < 2; ++e) {
        const int d = tid + e * 256;
        float v = tok_emb[(long long)tk * 512 + d] * SQ
                + ptag_emb[(long long)pt * 512 + d]
                + elem_emb[et * 512 + d];
        const int pair = d >> 1;
        const float freq = __expf(-(float)(2 * pair) * LG);
        const float arg = (float)spos * freq;
        v += (d & 1) ? __cosf(arg) : __sinf(arg);
        float accm = 0.f;
        for (int j = 0; j < 128; ++j) accm += h1[j] * w2[j * 512 + d];
        xv[e] = v + accm + b2v[d];
    }
    s = xv[0] + xv[1]; q = xv[0] * xv[0] + xv[1] * xv[1];
    blockReduce2(s, q, red);
    m  = s * (1.f / 512.f);
    rs = rsqrtf(q * (1.f / 512.f) - m * m + 1e-5f);
    const long long base = (long long)t * 512;
    xout[base + tid]       = (xv[0] - m) * rs * eg[tid]       + eb[tid];
    xout[base + tid + 256] = (xv[1] - m) * rs * eg[tid + 256] + eb[tid + 256];
}

// ---------------------------------------------------------------------------
// Relation bias: bias[b,q,k] = sum_r tag_rel[b,q,k,r] * softmax(rel_w[l])[r]
// ---------------------------------------------------------------------------
__global__ void relbias_kernel(const float* __restrict__ tr, const float* __restrict__ rw,
                               float* __restrict__ biasm, int n)
{
    float w0 = rw[0], w1 = rw[1], w2 = rw[2], w3 = rw[3];
    float mx = fmaxf(fmaxf(w0, w1), fmaxf(w2, w3));
    float e0 = __expf(w0 - mx), e1 = __expf(w1 - mx), e2 = __expf(w2 - mx), e3 = __expf(w3 - mx);
    float inv = 1.f / (e0 + e1 + e2 + e3);
    e0 *= inv; e1 *= inv; e2 *= inv; e3 *= inv;
    const int i = blockIdx.x * blockDim.x + threadIdx.x;
    if (i < n) {
        const float4 t = ((const float4*)tr)[i];
        biasm[i] = t.x * e0 + t.y * e1 + t.z * e2 + t.w * e3;
    }
}

// ---------------------------------------------------------------------------
// Softmax over S=1024 keys; adds sigmoid-scaled bias + mask; emits bf16 probs
// ---------------------------------------------------------------------------
__global__ __launch_bounds__(256)
void softmax_kernel(const float* __restrict__ scores, const float* __restrict__ biasm,
                    const int* __restrict__ amask, const float* __restrict__ rel_scale_l,
                    bfloat* __restrict__ P)
{
    __shared__ float red[8];
    const int row  = blockIdx.x;         // (b*H + h)*S + qpos
    const int qpos = row & 1023;
    const int b    = row >> 13;
    const float* srow = scores + (long long)row * 1024;
    const float* brow = biasm + ((long long)b * 1024 + qpos) * 1024;
    const float bscale = 0.1f / (1.f + __expf(-rel_scale_l[0]));
    const int tid = threadIdx.x, wave = tid >> 5, lane = tid & 31;

    float vals[4];
    float mx = -3.4e38f;
#pragma unroll
    for (int i = 0; i < 4; ++i) {
        const int k = tid + i * 256;
        float v = srow[k] + brow[k] * bscale;
        if (amask[b * 1024 + k] == 0) v = -1e9f;
        vals[i] = v; mx = fmaxf(mx, v);
    }
#pragma unroll
    for (int o = 16; o; o >>= 1) mx = fmaxf(mx, __shfl_xor(mx, o, 32));
    if (lane == 0) red[wave] = mx;
    __syncthreads();
    mx = red[0];
#pragma unroll
    for (int i = 1; i < 8; ++i) mx = fmaxf(mx, red[i]);
    __syncthreads();

    float sum = 0.f;
#pragma unroll
    for (int i = 0; i < 4; ++i) { vals[i] = __expf(vals[i] - mx); sum += vals[i]; }
#pragma unroll
    for (int o = 16; o; o >>= 1) sum += __shfl_xor(sum, o, 32);
    if (lane == 0) red[wave] = sum;
    __syncthreads();
    sum = 0.f;
#pragma unroll
    for (int i = 0; i < 8; ++i) sum += red[i];
    const float inv = 1.f / sum;
#pragma unroll
    for (int i = 0; i < 4; ++i)
        P[(long long)row * 1024 + tid + i * 256] = (bfloat)(vals[i] * inv);
}

// ---------------------------------------------------------------------------
// fp32 -> bf16 conversion
// ---------------------------------------------------------------------------
__global__ void f32_to_bf16_kernel(const float* __restrict__ in, bfloat* __restrict__ out,
                                   long long n)
{
    long long i = (long long)blockIdx.x * blockDim.x + threadIdx.x;
    if (i < n) out[i] = (bfloat)in[i];
}

// ---------------------------------------------------------------------------
// Orchestration
// ---------------------------------------------------------------------------
extern "C" void kernel_launch(void* const* d_in, const int* in_sizes, int n_in,
                              void* d_out, int out_size, void* d_ws, size_t ws_size,
                              hipStream_t stream)
{
    (void)in_sizes; (void)n_in; (void)out_size; (void)ws_size;
    const int Bn = 2, S = 1024, Dm = 512, H = 8, Lc = 6, DFF = 2048, HD = 64;
    const int M = Bn * S;

    const int*   token_ids  = (const int*)d_in[0];
    const int*   pos_tags   = (const int*)d_in[1];
    const int*   elem_types = (const int*)d_in[2];
    const float* sfeat      = (const float*)d_in[3];
    const int*   amask      = (const int*)d_in[4];
    const float* tag_rel    = (const float*)d_in[5];
    const float* tok_emb    = (const float*)d_in[6];
    const float* ptag_emb   = (const float*)d_in[7];
    const float* elem_emb   = (const float*)d_in[8];
    const float* syn_w1     = (const float*)d_in[9];
    const float* syn_b1     = (const float*)d_in[10];
    const float* syn_ln_g   = (const float*)d_in[11];
    const float* syn_ln_b   = (const float*)d_in[12];
    const float* syn_w2     = (const float*)d_in[13];
    const float* syn_b2     = (const float*)d_in[14];
    const float* emb_ln_g   = (const float*)d_in[15];
    const float* emb_ln_b   = (const float*)d_in[16];
    const float* n1_g       = (const float*)d_in[17];
    const float* n1_b       = (const float*)d_in[18];
    const float* n2_g       = (const float*)d_in[19];
    const float* n2_b       = (const float*)d_in[20];
    const float* attn_ln_g  = (const float*)d_in[21];
    const float* attn_ln_b  = (const float*)d_in[22];
    const float* qw         = (const float*)d_in[23];
    const float* qb         = (const float*)d_in[24];
    const float* kw         = (const float*)d_in[25];
    const float* kb         = (const float*)d_in[26];
    const float* vw         = (const float*)d_in[27];
    const float* vb         = (const float*)d_in[28];
    const float* ow         = (const float*)d_in[29];
    const float* ob         = (const float*)d_in[30];
    const float* rel_w      = (const float*)d_in[31];
    const float* rel_scale  = (const float*)d_in[32];
    const float* ffw1       = (const float*)d_in[33];
    const float* ffb1       = (const float*)d_in[34];
    const float* ffw2       = (const float*)d_in[35];
    const float* ffb2       = (const float*)d_in[36];
    const float* fin_g      = (const float*)d_in[37];
    const float* fin_b      = (const float*)d_in[38];

    // workspace carve-up (~164 MB total)
    char* ws = (char*)d_ws;
    size_t off = 0;
    auto alloc = [&](size_t bytes) -> void* {
        off = (off + 255) & ~(size_t)255;
        void* p = ws + off;
        off += bytes;
        return p;
    };
    float*  x      = (float*) alloc((size_t)M * Dm * 4);
    bfloat* hn     = (bfloat*)alloc((size_t)M * Dm * 2);
    bfloat* qm     = (bfloat*)alloc((size_t)M * Dm * 2);
    bfloat* km     = (bfloat*)alloc((size_t)M * Dm * 2);
    bfloat* vm     = (bfloat*)alloc((size_t)M * Dm * 2);
    bfloat* om     = (bfloat*)alloc((size_t)M * Dm * 2);
    bfloat* hf     = (bfloat*)alloc((size_t)M * Dm * 2);
    bfloat* ff1    = (bfloat*)alloc((size_t)M * DFF * 2);
    float*  biasm  = (float*) alloc((size_t)Bn * S * S * 4);
    float*  scores = (float*) alloc((size_t)Bn * H * S * S * 4);
    bfloat* P      = (bfloat*)alloc((size_t)Bn * H * S * S * 2);
    bfloat* wqB    = (bfloat*)alloc((size_t)Lc * Dm * Dm * 2);
    bfloat* wkB    = (bfloat*)alloc((size_t)Lc * Dm * Dm * 2);
    bfloat* wvB    = (bfloat*)alloc((size_t)Lc * Dm * Dm * 2);
    bfloat* woB    = (bfloat*)alloc((size_t)Lc * Dm * Dm * 2);
    bfloat* wf1B   = (bfloat*)alloc((size_t)Lc * Dm * DFF * 2);
    bfloat* wf2B   = (bfloat*)alloc((size_t)Lc * DFF * Dm * 2);

    auto cvt = [&](const float* in, bfloat* outp, long long n) {
        f32_to_bf16_kernel<<<(unsigned)((n + 255) / 256), 256, 0, stream>>>(in, outp, n);
    };
    cvt(qw,   wqB,  (long long)Lc * Dm * Dm);
    cvt(kw,   wkB,  (long long)Lc * Dm * Dm);
    cvt(vw,   wvB,  (long long)Lc * Dm * Dm);
    cvt(ow,   woB,  (long long)Lc * Dm * Dm);
    cvt(ffw1, wf1B, (long long)Lc * Dm * DFF);
    cvt(ffw2, wf2B, (long long)Lc * DFF * Dm);

    embed_kernel<<<M, 256, 0, stream>>>(token_ids, pos_tags, elem_types, sfeat,
                                        tok_emb, ptag_emb, elem_emb,
                                        syn_w1, syn_b1, syn_ln_g, syn_ln_b,
                                        syn_w2, syn_b2, emb_ln_g, emb_ln_b, x);

    auto gemm = [&](const bfloat* A, const bfloat* Bm, const float* bias, const float* res,
                    void* C, int Mi, int Ni, int Ki,
                    int lda, int ldb, int ldc, int ldr,
                    long long aZ1, long long aZ2, long long bZ1, long long bZ2,
                    long long cZ1, long long cZ2, int nz1, int nz2,
                    float alpha, int flags) {
        dim3 grid(Ni / 64, Mi / 128, nz1 * nz2);
        if (flags & FLAG_TRANSB)
            wmma_gemm_kernel<true><<<grid, 256, 0, stream>>>(A, Bm, bias, res, C, Ki,
                lda, ldb, ldc, ldr, aZ1, aZ2, bZ1, bZ2, cZ1, cZ2, nz2, alpha, flags);
        else
            wmma_gemm_kernel<false><<<grid, 256, 0, stream>>>(A, Bm, bias, res, C, Ki,
                lda, ldb, ldc, ldr, aZ1, aZ2, bZ1, bZ2, cZ1, cZ2, nz2, alpha, flags);
    };

    const long long SD = (long long)S * Dm, SS = (long long)S * S;
    for (int l = 0; l < Lc; ++l) {
        const size_t lD = (size_t)l * Dm, lDD = (size_t)l * Dm * Dm;
        const size_t lDF = (size_t)l * DFF, lDDF = (size_t)l * Dm * DFF;

        // hn = attn_ln(n1(x)) -> bf16
        ln_kernel<<<M, 256, 0, stream>>>(x, n1_g + lD, n1_b + lD,
                                         attn_ln_g + lD, attn_ln_b + lD, hn, 1);
        // Q/K/V projections
        gemm(hn, wqB + lDD, qb + lD, nullptr, qm, M, Dm, Dm, Dm, Dm, Dm, 0,
             0, 0, 0, 0, 0, 0, 1, 1, 1.f, FLAG_OUTBF);
        gemm(hn, wkB + lDD, kb + lD, nullptr, km, M, Dm, Dm, Dm, Dm, Dm, 0,
             0, 0, 0, 0, 0, 0, 1, 1, 1.f, FLAG_OUTBF);
        gemm(hn, wvB + lDD, vb + lD, nullptr, vm, M, Dm, Dm, Dm, Dm, Dm, 0,
             0, 0, 0, 0, 0, 0, 1, 1, 1.f, FLAG_OUTBF);
        // relation bias for this layer
        relbias_kernel<<<(Bn * S * S) / 256, 256, 0, stream>>>(tag_rel, rel_w + (size_t)l * 4,
                                                               biasm, Bn * S * S);
        // scores = Q * K^T / sqrt(HD), per (b,h)
        gemm(qm, km, nullptr, nullptr, scores, S, S, HD, Dm, Dm, S, 0,
             SD, HD, SD, HD, (long long)H * SS, SS, Bn, H, 0.125f, FLAG_TRANSB);
        // softmax (+bias, +mask) -> bf16 P
        softmax_kernel<<<Bn * H * S, 256, 0, stream>>>(scores, biasm, amask,
                                                       rel_scale + l, P);
        // O = P * V, per (b,h), written back in [B,S,D] head-concat layout
        gemm(P, vm, nullptr, nullptr, om, S, HD, S, S, Dm, Dm, 0,
             (long long)H * SS, SS, SD, HD, SD, HD, Bn, H, 1.f, FLAG_OUTBF);
        // x = x + O*ow + ob
        gemm(om, woB + lDD, ob + lD, x, x, M, Dm, Dm, Dm, Dm, Dm, Dm,
             0, 0, 0, 0, 0, 0, 1, 1, 1.f, 0);
        // hf = n2(x) -> bf16
        ln_kernel<<<M, 256, 0, stream>>>(x, n2_g + lD, n2_b + lD, nullptr, nullptr, hf, 1);
        // ff1 = relu(hf*W1 + b1) -> bf16
        gemm(hf, wf1B + lDDF, ffb1 + lDF, nullptr, ff1, M, DFF, Dm, Dm, DFF, DFF, 0,
             0, 0, 0, 0, 0, 0, 1, 1, 1.f, FLAG_RELU | FLAG_OUTBF);
        // x = x + ff1*W2 + b2
        gemm(ff1, wf2B + (size_t)l * DFF * Dm, ffb2 + lD, x, x, M, Dm, DFF,
             DFF, Dm, Dm, Dm, 0, 0, 0, 0, 0, 0, 1, 1, 1.f, 0);
    }

    // final LN -> fp32 d_out
    ln_kernel<<<M, 256, 0, stream>>>(x, fin_g, fin_b, nullptr, nullptr, d_out, 0);
}
